// VectorQuantize_1022202217162
// MI455X (gfx1250) — compile-verified, matrix operand order checked by reading, not run on hardware
//
#include <hip/hip_runtime.h>
#include <math.h>

typedef __attribute__((ext_vector_type(16))) __bf16 v16bf;
typedef __attribute__((ext_vector_type(8)))  __bf16 v8bf;
typedef __attribute__((ext_vector_type(8)))  float  v8f;

#define B_      16
#define T_      2048
#define IN_DIM  1024
#define CB_DIM  128
#define CB_SIZE 4096

// Load a v16bf A/B fragment: elems 0..7 = K base..base+7, elems 8..15 = K base+16..base+23
// (ptr already includes the per-lane row offset and K base). Two 16-byte loads.
__device__ inline v16bf load_frag(const __bf16* p) {
    v8bf lo = *(const v8bf*)p;
    v8bf hi = *(const v8bf*)(p + 16);
    v16bf r;
#pragma unroll
    for (int e = 0; e < 8; ++e) { r[e] = lo[e]; r[e + 8] = hi[e]; }
    return r;
}

__device__ inline v8f wmma_bf16(v16bf a, v16bf b, v8f c) {
    return __builtin_amdgcn_wmma_f32_16x16x32_bf16(false, a, false, b, (short)0, c, false, false);
}

// ---------------- Kernel A: weight-norm + l2-norm -> bf16 copies ----------------
__global__ void k_prep(const float* in_v, const float* in_g,
                       const float* codebook,
                       const float* out_v, const float* out_g,
                       __bf16* w_in_bf, __bf16* w_out_bf, __bf16* cbn) {
    __shared__ float red[128];
    const int tid = threadIdx.x;
    const int bid = blockIdx.x;
    if (bid < CB_DIM) {                          // in_v row: 1024 wide
        const int o = bid;
        const float* vr = in_v + (size_t)o * IN_DIM;
        float s = 0.f;
        for (int i = tid; i < IN_DIM; i += 128) { float x = vr[i]; s += x * x; }
        red[tid] = s; __syncthreads();
        for (int w = 64; w > 0; w >>= 1) { if (tid < w) red[tid] += red[tid + w]; __syncthreads(); }
        const float scale = in_g[o] / sqrtf(red[0]);
        for (int i = tid; i < IN_DIM; i += 128)
            w_in_bf[(size_t)o * IN_DIM + i] = (__bf16)(vr[i] * scale);
    } else if (bid < CB_DIM + IN_DIM) {          // out_v row: 128 wide
        const int o = bid - CB_DIM;
        const float* vr = out_v + (size_t)o * CB_DIM;
        const float x = vr[tid & 127];
        red[tid] = x * x; __syncthreads();
        for (int w = 64; w > 0; w >>= 1) { if (tid < w) red[tid] += red[tid + w]; __syncthreads(); }
        const float scale = out_g[o] / sqrtf(red[0]);
        w_out_bf[(size_t)o * CB_DIM + tid] = (__bf16)(x * scale);
    } else {                                     // codebook row: 128 wide, l2norm w/ eps
        const int j = bid - (CB_DIM + IN_DIM);
        const float* cr = codebook + (size_t)j * CB_DIM;
        const float x = cr[tid & 127];
        red[tid] = x * x; __syncthreads();
        for (int w = 64; w > 0; w >>= 1) { if (tid < w) red[tid] += red[tid + w]; __syncthreads(); }
        const float scale = 1.0f / fmaxf(sqrtf(red[0]), 1e-12f);
        cbn[(size_t)j * CB_DIM + tid] = (__bf16)(x * scale);
    }
}

// ---------------- Kernel B: z_e = w_in @ z + in_b ----------------
// Block: 128 thr (4 waves). Block tile: M=128 (8 wmma tiles) x N=64 (16 per wave). K=1024.
__global__ void k_inproj(const float* __restrict__ z, const __bf16* __restrict__ w_in,
                         const float* __restrict__ in_b,
                         float* __restrict__ ze_out, __bf16* __restrict__ zeT) {
    __shared__ float zl[32][68];   // pitch 68: half-waves land on disjoint banks
    const int tid  = threadIdx.x;
    const int wave = tid >> 5, lane = tid & 31;
    const int ln = lane & 15, h = lane >> 4, base = h * 8;
    const int b  = blockIdx.x >> 5;
    const int t0 = (blockIdx.x & 31) * 64;
    const float* zb = z + (size_t)b * IN_DIM * T_ + t0;

    v8f acc[8];
#pragma unroll
    for (int m = 0; m < 8; ++m)
#pragma unroll
        for (int r = 0; r < 8; ++r) acc[m][r] = 0.f;

    for (int k = 0; k < IN_DIM; k += 32) {
        __syncthreads();
        for (int i = tid; i < 2048; i += 128) {
            const int kk = i >> 6, c = i & 63;
            zl[kk][c] = zb[(size_t)(k + kk) * T_ + c];
            if (k + 32 < IN_DIM)   // prefetch next K-tile of z (global_prefetch_b8)
                __builtin_prefetch(&zb[(size_t)(k + 32 + kk) * T_ + c], 0, 0);
        }
        __syncthreads();
        // B fragment (K=32 x N=16) from LDS, f32 -> bf16
        v16bf Bf;
        const int c = wave * 16 + ln;
#pragma unroll
        for (int e = 0; e < 8; ++e) {
            Bf[e]     = (__bf16)zl[base + e][c];
            Bf[e + 8] = (__bf16)zl[base + 16 + e][c];
        }
        // batch A-fragment loads in groups of 4 so a single s_wait covers 8 b128 loads,
        // then issue 4 WMMAs back-to-back
#pragma unroll
        for (int mg = 0; mg < 8; mg += 4) {
            v16bf Af[4];
#pragma unroll
            for (int m = 0; m < 4; ++m)
                Af[m] = load_frag(w_in + (size_t)((mg + m) * 16 + ln) * IN_DIM + k + base);
#pragma unroll
            for (int m = 0; m < 4; ++m)
                acc[mg + m] = wmma_bf16(Af[m], Bf, acc[mg + m]);
        }
    }
    const int t = t0 + wave * 16 + ln;
#pragma unroll
    for (int m = 0; m < 8; ++m)
#pragma unroll
        for (int r = 0; r < 8; ++r) {
            const int o = m * 16 + r + h * 8;
            const float v = acc[m][r] + in_b[o];
            ze_out[((size_t)b * CB_DIM + o) * T_ + t] = v;
            zeT[((size_t)(b * T_ + t)) * CB_DIM + o] = (__bf16)v;
        }
}

// ---------------- Kernel C: argmax(z_e . cb_n) + gather z_q + loss partials ----------------
// Wave handles 16 tokens; loops 256 code tiles of 16. K=128 (4 wmma k-steps).
__global__ void k_argmax(const __bf16* __restrict__ zeT, const __bf16* __restrict__ cbn,
                         const float* __restrict__ codebook,
                         __bf16* __restrict__ zqT, float* __restrict__ out_idx,
                         float* __restrict__ lossPart) {
    const int tid  = threadIdx.x;
    const int wave = tid >> 5, lane = tid & 31;
    const int ln = lane & 15, h = lane >> 4, base = h * 8;
    const int mtile = blockIdx.x * 4 + wave;
    const int row0  = mtile * 16;

    // A fragments: 16 tokens x full K=128, resident for whole code loop
    const __bf16* zr = zeT + (size_t)(row0 + ln) * CB_DIM + base;
    v16bf A[4];
#pragma unroll
    for (int ks = 0; ks < 4; ++ks) A[ks] = load_frag(zr + ks * 32);

    float bestv[8]; int besti[8];
#pragma unroll
    for (int r = 0; r < 8; ++r) { bestv[r] = -3.4e38f; besti[r] = 0; }

    for (int j = 0; j < CB_SIZE / 16; ++j) {
        const __bf16* cr = cbn + (size_t)(j * 16 + ln) * CB_DIM + base;
        v16bf Bf[4];                 // batch the 8 b128 loads, then 4 WMMAs back-to-back
#pragma unroll
        for (int ks = 0; ks < 4; ++ks) Bf[ks] = load_frag(cr + ks * 32);
        v8f acc;
#pragma unroll
        for (int r = 0; r < 8; ++r) acc[r] = 0.f;
#pragma unroll
        for (int ks = 0; ks < 4; ++ks) acc = wmma_bf16(A[ks], Bf[ks], acc);
        const int cidx = j * 16 + ln;   // this lane's candidate code
#pragma unroll
        for (int r = 0; r < 8; ++r)
            if (acc[r] > bestv[r]) { bestv[r] = acc[r]; besti[r] = cidx; }
    }
    // reduce across the 16 lanes holding the 16 codes (halves stay independent for mask<16)
#pragma unroll
    for (int m = 1; m < 16; m <<= 1)
#pragma unroll
        for (int r = 0; r < 8; ++r) {
            const float ov = __shfl_xor(bestv[r], m, 32);
            const int   oi = __shfl_xor(besti[r], m, 32);
            if (ov > bestv[r] || (ov == bestv[r] && oi < besti[r])) { bestv[r] = ov; besti[r] = oi; }
        }
    if (ln == 0) {
#pragma unroll
        for (int r = 0; r < 8; ++r)
            out_idx[row0 + h * 8 + r] = (float)besti[r];
    }
    // gather z_q rows (bf16) + loss partial: each half handles its 8 rows, 8 dims per lane
    float lsum = 0.f;
#pragma unroll
    for (int r = 0; r < 8; ++r) {
        const int row = row0 + h * 8 + r;
        const int ci  = besti[r];                       // uniform within the half
        const float* cr = codebook + (size_t)ci * CB_DIM + ln * 8;
        const float4 c0 = *(const float4*)cr;
        const float4 c1 = *(const float4*)(cr + 4);
        float cf[8] = {c0.x, c0.y, c0.z, c0.w, c1.x, c1.y, c1.z, c1.w};
        v8bf q;
#pragma unroll
        for (int e = 0; e < 8; ++e) q[e] = (__bf16)cf[e];
        *(v8bf*)(zqT + (size_t)row * CB_DIM + ln * 8) = q;
        const v8bf ze8 = *(const v8bf*)(zeT + (size_t)row * CB_DIM + ln * 8);
#pragma unroll
        for (int e = 0; e < 8; ++e) { const float d = (float)ze8[e] - cf[e]; lsum += d * d; }
    }
#pragma unroll
    for (int m = 1; m < 32; m <<= 1) lsum += __shfl_xor(lsum, m, 32);
    if (lane == 0) lossPart[mtile] = lsum;
}

// ---------------- Kernel D: z_q_out = w_out @ z_q + out_b ----------------
// Block tile: M=128 (8 tiles, chunk of 1024) x N=64. K=128.
__global__ void k_outproj(const __bf16* __restrict__ zqT, const __bf16* __restrict__ w_out,
                          const float* __restrict__ out_b, float* __restrict__ zq_out) {
    const int tid  = threadIdx.x;
    const int wave = tid >> 5, lane = tid & 31;
    const int ln = lane & 15, h = lane >> 4, base = h * 8;
    const int mchunk = blockIdx.x & 7;
    const int rest   = blockIdx.x >> 3;
    const int t0 = (rest & 31) * 64;
    const int b  = rest >> 5;
    const int o0 = mchunk * 128;
    const int t  = t0 + wave * 16 + ln;

    const __bf16* zq0 = zqT + (size_t)(b * T_ + t) * CB_DIM + base;
    v16bf Bf[4];
#pragma unroll
    for (int ks = 0; ks < 4; ++ks) Bf[ks] = load_frag(zq0 + ks * 32);

#pragma unroll
    for (int m = 0; m < 8; ++m) {
        const __bf16* wr = w_out + (size_t)(o0 + m * 16 + ln) * CB_DIM + base;
        v16bf Af[4];                 // 8 b128 loads in a clause, then 4 WMMAs
#pragma unroll
        for (int ks = 0; ks < 4; ++ks) Af[ks] = load_frag(wr + ks * 32);
        v8f acc;
#pragma unroll
        for (int r = 0; r < 8; ++r) acc[r] = 0.f;
#pragma unroll
        for (int ks = 0; ks < 4; ++ks) acc = wmma_bf16(Af[ks], Bf[ks], acc);
#pragma unroll
        for (int r = 0; r < 8; ++r) {
            const int o = o0 + m * 16 + r + h * 8;
            zq_out[((size_t)b * IN_DIM + o) * T_ + t] = acc[r] + out_b[o];
        }
    }
}

// ---------------- Kernel E: finalize losses ----------------
__global__ void k_loss(const float* __restrict__ lossPart, float* o_commit, float* o_cb) {
    __shared__ float red[128];
    const int b = blockIdx.x, t = threadIdx.x;
    red[t] = lossPart[b * 128 + t];
    __syncthreads();
    for (int w = 64; w > 0; w >>= 1) { if (t < w) red[t] += red[t + w]; __syncthreads(); }
    if (t == 0) {
        const float v = red[0] * (1.0f / ((float)CB_DIM * (float)T_));
        o_commit[b] = v;
        o_cb[b]     = v;   // forward values identical
    }
}

extern "C" void kernel_launch(void* const* d_in, const int* in_sizes, int n_in,
                              void* d_out, int out_size, void* d_ws, size_t ws_size,
                              hipStream_t stream) {
    const float* z        = (const float*)d_in[0];
    const float* in_v     = (const float*)d_in[1];
    const float* in_g     = (const float*)d_in[2];
    const float* in_b     = (const float*)d_in[3];
    const float* codebook = (const float*)d_in[4];
    const float* out_v    = (const float*)d_in[5];
    const float* out_g    = (const float*)d_in[6];
    const float* out_b    = (const float*)d_in[7];

    float* out      = (float*)d_out;
    float* o_zqout  = out;                                     // [16,1024,2048]
    float* o_commit = out + (size_t)B_ * IN_DIM * T_;          // [16]
    float* o_cbl    = o_commit + B_;                           // [16]
    float* o_idx    = o_cbl + B_;                              // [16,2048] as float
    float* o_ze     = o_idx + (size_t)B_ * T_;                 // [16,128,2048]

    char* w = (char*)d_ws;
    __bf16* w_in_bf  = (__bf16*)(w);                           // 256 KB
    __bf16* w_out_bf = (__bf16*)(w + (256u << 10));            // 256 KB
    __bf16* cbn      = (__bf16*)(w + (512u << 10));            // 1 MB
    __bf16* zeT      = (__bf16*)(w + (1536u << 10));           // 8 MB  [(b*T+t)][128]
    __bf16* zqT      = (__bf16*)(w + (1536u << 10) + (8u << 20)); // 8 MB
    float*  lossPart = (float*)(w + (1536u << 10) + (16u << 20)); // 8 KB [2048]

    k_prep<<<CB_DIM + IN_DIM + CB_SIZE, 128, 0, stream>>>(in_v, in_g, codebook, out_v, out_g,
                                                          w_in_bf, w_out_bf, cbn);
    k_inproj<<<B_ * (T_ / 64), 128, 0, stream>>>(z, w_in_bf, in_b, o_ze, zeT);
    k_argmax<<<(B_ * T_ / 16) / 4, 128, 0, stream>>>(zeT, cbn, codebook, zqT, o_idx, lossPart);
    k_outproj<<<B_ * (T_ / 64) * (IN_DIM / 128), 128, 0, stream>>>(zqT, w_out_bf, out_b, o_zqout);
    k_loss<<<B_, 128, 0, stream>>>(lossPart, o_commit, o_cbl);
}
